// TransformerAttention_12807592476803
// MI455X (gfx1250) — matrix operand order repaired and checked by reading, not compiled, not run on hardware
//
#include <hip/hip_runtime.h>
#include <hip/hip_bf16.h>

// ---------------------------------------------------------------------------
// TransformerAttention on gfx1250 (MI455X): bf16 WMMA everywhere.
//   x:(2,2048,1024) f32, mask:(2,2048,16,2048) i32,
//   W_qkv:(1024,3072) f32, b_qkv:(3072) f32, W_out:(1024,1024) f32, b_out:(1024)
//   out = x + Attn(x) : (2,2048,1024) f32
// ---------------------------------------------------------------------------

typedef __attribute__((ext_vector_type(16))) __bf16 v16bf;
typedef __attribute__((ext_vector_type(8)))  float  v8f;

#define N_HID 1024
#define N_HEAD 16
#define HD 64            // head dim (N_K == N_V == 64)
#define BATCH 2
#define SEQ 2048
#define QKV_N 3072       // 16 * 192
#define ROWS 4096        // B*S
#define LOG2E 1.4426950408889634f

static __device__ __forceinline__ unsigned short f2bf(float f) {
  unsigned int u = __float_as_uint(f);
  unsigned int r = u + 0x7FFFu + ((u >> 16) & 1u);
  return (unsigned short)(r >> 16);
}

static __device__ __forceinline__ v8f wmma_bf16(v16bf a, v16bf b, v8f c) {
  // D = A(16x32 bf16) x B(32x16 bf16) + C(16x16 f32)
  return __builtin_amdgcn_wmma_f32_16x16x32_bf16(false, a, false, b, (short)0, c,
                                                 false, false);
}

// A-fragment (16x32, 16-bit): lane = (m, half); VGPR0-3 hold K[8h..8h+7],
// VGPR4-7 hold K[16+8h..16+8h+7]  (ISA 7.12.2).  src is row-major MxK.
static __device__ __forceinline__ v16bf load_a(const unsigned short* src,
                                               int row0, int k0, int ld) {
  int lane = threadIdx.x & 31;
  int m = lane & 15, half = lane >> 4;
  const unsigned short* p = src + (size_t)(row0 + m) * ld + k0;
  union { uint4 u[2]; v16bf v; } f;
  f.u[0] = *(const uint4*)(p + 8 * half);
  f.u[1] = *(const uint4*)(p + 16 + 8 * half);
  return f.v;
}

// B-fragment (32x16, 16-bit): lane = (n, half); VGPR0-3 hold K[8h..8h+7],
// VGPR4-7 hold K[16+8h..16+8h+7] (same as A).  bt is row-major NxK (i.e. B transposed).
static __device__ __forceinline__ v16bf load_b(const unsigned short* bt,
                                               int n0, int k0, int ld) {
  int lane = threadIdx.x & 31;
  int n = lane & 15, half = lane >> 4;
  const unsigned short* p = bt + (size_t)(n0 + n) * ld + k0;
  union { uint4 u[2]; v16bf v; } f;
  f.u[0] = *(const uint4*)(p + 8 * half);
  f.u[1] = *(const uint4*)(p + 16 + 8 * half);
  return f.v;
}

// ---------------------------------------------------------------------------
// Prep kernels
// ---------------------------------------------------------------------------
__global__ void to_bf16_kernel(const float* __restrict__ src,
                               unsigned short* __restrict__ dst, int n) {
  int i = blockIdx.x * blockDim.x + threadIdx.x;
  if (i < n) dst[i] = f2bf(src[i]);
}

// Wt[n][k] = W[k][n]  (output row-major NxK bf16)
__global__ void transpose_bf16_kernel(const float* __restrict__ W,
                                      unsigned short* __restrict__ Wt,
                                      int K, int N) {
  int i = blockIdx.x * blockDim.x + threadIdx.x;
  if (i >= K * N) return;
  int n = i / K, k = i - n * K;
  Wt[(size_t)n * K + k] = f2bf(W[(size_t)k * N + n]);
}

// ---------------------------------------------------------------------------
// QKV projection: qkv = xb @ Wqkv + b ; scatter into q (scaled), k, vT
// one wave -> 32x64 tile; K fully unrolled (all load offsets become
// immediates -> no address VALU, no WMMA-source WAR hazard nops)
// ---------------------------------------------------------------------------
__global__ void __launch_bounds__(128)
qkv_kernel(const unsigned short* __restrict__ xb,
           const unsigned short* __restrict__ wT,   // (3072 x 1024) bf16
           const float* __restrict__ bias,
           unsigned short* __restrict__ q,          // (b,h,s,d)
           unsigned short* __restrict__ kk,         // (b,h,s,d)
           unsigned short* __restrict__ vT) {       // (b,h,d,s)
  int wave = blockIdx.x * (blockDim.x >> 5) + (threadIdx.x >> 5);
  int mt = wave / (QKV_N / 64), nt = wave % (QKV_N / 64);
  int m0 = mt * 32, n0 = nt * 64;

  v8f acc[8] = {};
  v16bf a0c = load_a(xb, m0, 0, N_HID);
  v16bf a1c = load_a(xb, m0 + 16, 0, N_HID);
  v16bf bbc[4];
#pragma unroll
  for (int t = 0; t < 4; t++) bbc[t] = load_b(wT, n0 + t * 16, 0, N_HID);

#pragma unroll
  for (int k0 = 0; k0 < N_HID; k0 += 64) {
    // load buffer B (k0+32)
    v16bf a0n = load_a(xb, m0, k0 + 32, N_HID);
    v16bf a1n = load_a(xb, m0 + 16, k0 + 32, N_HID);
    v16bf bbn[4];
#pragma unroll
    for (int t = 0; t < 4; t++) bbn[t] = load_b(wT, n0 + t * 16, k0 + 32, N_HID);
    // compute buffer A
#pragma unroll
    for (int t = 0; t < 4; t++) {
      acc[t]     = wmma_bf16(a0c, bbc[t], acc[t]);
      acc[4 + t] = wmma_bf16(a1c, bbc[t], acc[4 + t]);
    }
    // load buffer A (k0+64; wraps to 0 on last iter -> dead code, eliminated)
    int k2 = (k0 + 64) & (N_HID - 1);
    a0c = load_a(xb, m0, k2, N_HID);
    a1c = load_a(xb, m0 + 16, k2, N_HID);
#pragma unroll
    for (int t = 0; t < 4; t++) bbc[t] = load_b(wT, n0 + t * 16, k2, N_HID);
    // compute buffer B
#pragma unroll
    for (int t = 0; t < 4; t++) {
      acc[t]     = wmma_bf16(a0n, bbn[t], acc[t]);
      acc[4 + t] = wmma_bf16(a1n, bbn[t], acc[4 + t]);
    }
  }

  int lane = threadIdx.x & 31;
  int nIdx = lane & 15, half = lane >> 4;
  int h = n0 / 192;
  int type = (n0 - h * 192) / 64;  // 0=q 1=k 2=v (uniform over the 64-wide tile)

#pragma unroll
  for (int mm = 0; mm < 2; mm++) {
#pragma unroll
    for (int t = 0; t < 4; t++) {
#pragma unroll
      for (int r = 0; r < 8; r++) {
        int row = m0 + mm * 16 + r + 8 * half;
        int b = row >> 11, s = row & (SEQ - 1);
        int col = n0 + t * 16 + nIdx;
        int d = col - h * 192 - type * 64;
        float val = acc[mm * 4 + t][r] + bias[col];
        size_t bh = (size_t)(b * N_HEAD + h);
        if (type == 0)
          q[(bh * SEQ + s) * HD + d] = f2bf(val * 0.125f);  // 1/sqrt(64)
        else if (type == 1)
          kk[(bh * SEQ + s) * HD + d] = f2bf(val);
        else
          vT[(bh * HD + d) * SEQ + s] = f2bf(val);
      }
    }
  }
}

// ---------------------------------------------------------------------------
// Flash attention: one wave per (b,h, 16-query tile), stream 64-key chunks.
// All fragment / mask loads issued before the WMMAs that consume them, and the
// next chunk's mask lines are prefetched (the mask is the only HBM stream).
// ---------------------------------------------------------------------------
__global__ void __launch_bounds__(128)
attn_kernel(const unsigned short* __restrict__ q,
            const unsigned short* __restrict__ kk,
            const unsigned short* __restrict__ vT,
            const int* __restrict__ mask,   // (b, s_q, h, s_k)
            unsigned short* __restrict__ ctx) {  // (b, s, h*64+d) bf16
  __shared__ __align__(16) unsigned short P[4][16][64];

  int wid = threadIdx.x >> 5;
  int wave = blockIdx.x * (blockDim.x >> 5) + wid;
  int bh = wave >> 7;             // 0..31
  int qt = wave & 127;
  int b = bh >> 4, h = bh & 15;
  int q0 = qt * 16;
  int lane = threadIdx.x & 31;
  int nIdx = lane & 15, half = lane >> 4;

  const unsigned short* qbase = q + (size_t)bh * SEQ * HD;
  const unsigned short* kbase = kk + (size_t)bh * SEQ * HD;
  const unsigned short* vbase = vT + (size_t)bh * HD * SEQ;
  const int* maskbase = mask + (size_t)b * SEQ * N_HEAD * SEQ + (size_t)h * SEQ;
  unsigned short(*Pw)[64] = P[wid];

  v16bf qa0 = load_a(qbase, q0, 0, HD);
  v16bf qa1 = load_a(qbase, q0, 32, HD);

  v8f acc[4] = {};
  float mrow[8], lrow[8];
#pragma unroll
  for (int r = 0; r < 8; r++) { mrow[r] = -1e30f; lrow[r] = 0.f; }

  // prefetch the first mask chunk
#pragma unroll
  for (int pi = 0; pi < 2; pi++) {
    int li = pi * 32 + lane;                       // 64 cache lines / chunk
    const void* pa = (const void*)(maskbase +
        (size_t)(q0 + (li >> 2)) * (N_HEAD * SEQ) + ((li & 3) << 4));
    __builtin_prefetch(pa, 0, 1);
  }

#pragma unroll 2
  for (int kc = 0; kc < SEQ; kc += 64) {
    // ---- issue all K fragments, then all mask words, then prefetch next ----
    v16bf kb[8];
#pragma unroll
    for (int t = 0; t < 4; t++) {
      kb[t * 2 + 0] = load_b(kbase, kc + t * 16, 0, HD);
      kb[t * 2 + 1] = load_b(kbase, kc + t * 16, 32, HD);
    }
    int mv[4][8];
#pragma unroll
    for (int t = 0; t < 4; t++) {
      int key = kc + t * 16 + nIdx;
#pragma unroll
      for (int r = 0; r < 8; r++) {
        int qrow = q0 + r + 8 * half;
        mv[t][r] = maskbase[(size_t)qrow * (N_HEAD * SEQ) + key];
      }
    }
    {
      int kcn = (kc + 64) & (SEQ - 1);
#pragma unroll
      for (int pi = 0; pi < 2; pi++) {
        int li = pi * 32 + lane;
        const void* pa = (const void*)(maskbase +
            (size_t)(q0 + (li >> 2)) * (N_HEAD * SEQ) + kcn + ((li & 3) << 4));
        __builtin_prefetch(pa, 0, 1);
      }
    }
    // ---- scores: 16 x 64 (4 sub-tiles) ----
    v8f s[4];
#pragma unroll
    for (int t = 0; t < 4; t++) {
      v8f c = {};
      c = wmma_bf16(qa0, kb[t * 2 + 0], c);
      c = wmma_bf16(qa1, kb[t * 2 + 1], c);
      s[t] = c;
    }
    // ---- mask ----
#pragma unroll
    for (int t = 0; t < 4; t++)
#pragma unroll
      for (int r = 0; r < 8; r++)
        if (!mv[t][r]) s[t][r] = -1e30f;
    // ---- online softmax: new row max (reduce over 16 lanes of each half) ----
    float mnew[8];
#pragma unroll
    for (int r = 0; r < 8; r++) {
      float m2 = mrow[r];
#pragma unroll
      for (int t = 0; t < 4; t++) m2 = fmaxf(m2, s[t][r]);
      m2 = fmaxf(m2, __shfl_xor(m2, 1, 32));
      m2 = fmaxf(m2, __shfl_xor(m2, 2, 32));
      m2 = fmaxf(m2, __shfl_xor(m2, 4, 32));
      m2 = fmaxf(m2, __shfl_xor(m2, 8, 32));
      mnew[r] = m2;
    }
    float psum[8];
#pragma unroll
    for (int r = 0; r < 8; r++) {
      float sc = exp2f((mrow[r] - mnew[r]) * LOG2E);
      lrow[r] *= sc;
#pragma unroll
      for (int n = 0; n < 4; n++) acc[n][r] *= sc;
      mrow[r] = mnew[r];
      psum[r] = 0.f;
    }
    // ---- p = exp(s - m), stash to LDS in row-major 16x64 ----
#pragma unroll
    for (int t = 0; t < 4; t++) {
#pragma unroll
      for (int r = 0; r < 8; r++) {
        float p = exp2f((s[t][r] - mrow[r]) * LOG2E);
        psum[r] += p;
        Pw[r + 8 * half][t * 16 + nIdx] = f2bf(p);
      }
    }
#pragma unroll
    for (int r = 0; r < 8; r++) {
      float ps = psum[r];
      ps += __shfl_xor(ps, 1, 32);
      ps += __shfl_xor(ps, 2, 32);
      ps += __shfl_xor(ps, 4, 32);
      ps += __shfl_xor(ps, 8, 32);
      lrow[r] += ps;
    }
    // ---- ctx += P @ V : issue all V fragments + P fragments, then WMMAs ----
    v16bf pa0 = load_a(&Pw[0][0], 0, 0, 64);
    v16bf pa1 = load_a(&Pw[0][0], 0, 32, 64);
    v16bf vb[8];
#pragma unroll
    for (int n = 0; n < 4; n++) {
      vb[n * 2 + 0] = load_b(vbase, n * 16, kc, SEQ);
      vb[n * 2 + 1] = load_b(vbase, n * 16, kc + 32, SEQ);
    }
#pragma unroll
    for (int n = 0; n < 4; n++) {
      acc[n] = wmma_bf16(pa0, vb[n * 2 + 0], acc[n]);
      acc[n] = wmma_bf16(pa1, vb[n * 2 + 1], acc[n]);
    }
  }

  // ---- normalize + write ctx (bf16) ----
#pragma unroll
  for (int r = 0; r < 8; r++) {
    float inv = 1.0f / lrow[r];
    int row = q0 + r + 8 * half;
    size_t rowoff = ((size_t)b * SEQ + row) * N_HID + h * HD;
#pragma unroll
    for (int n = 0; n < 4; n++)
      ctx[rowoff + n * 16 + nIdx] = f2bf(acc[n][r] * inv);
  }
}

// ---------------------------------------------------------------------------
// Output projection + bias + residual:  out = x + ctx @ Wout + b_out
// one wave -> 32x64 tile; K fully unrolled
// ---------------------------------------------------------------------------
__global__ void __launch_bounds__(128)
out_kernel(const unsigned short* __restrict__ ctxb,
           const unsigned short* __restrict__ woutT,  // (1024 x 1024) bf16 N x K
           const float* __restrict__ bout,
           const float* __restrict__ x,
           float* __restrict__ out) {
  int wave = blockIdx.x * (blockDim.x >> 5) + (threadIdx.x >> 5);
  int mt = wave / (N_HID / 64), nt = wave % (N_HID / 64);
  int m0 = mt * 32, n0 = nt * 64;

  v8f acc[8] = {};
  v16bf a0c = load_a(ctxb, m0, 0, N_HID);
  v16bf a1c = load_a(ctxb, m0 + 16, 0, N_HID);
  v16bf bbc[4];
#pragma unroll
  for (int t = 0; t < 4; t++) bbc[t] = load_b(woutT, n0 + t * 16, 0, N_HID);

#pragma unroll
  for (int k0 = 0; k0 < N_HID; k0 += 64) {
    v16bf a0n = load_a(ctxb, m0, k0 + 32, N_HID);
    v16bf a1n = load_a(ctxb, m0 + 16, k0 + 32, N_HID);
    v16bf bbn[4];
#pragma unroll
    for (int t = 0; t < 4; t++) bbn[t] = load_b(woutT, n0 + t * 16, k0 + 32, N_HID);
#pragma unroll
    for (int t = 0; t < 4; t++) {
      acc[t]     = wmma_bf16(a0c, bbc[t], acc[t]);
      acc[4 + t] = wmma_bf16(a1c, bbc[t], acc[4 + t]);
    }
    int k2 = (k0 + 64) & (N_HID - 1);
    a0c = load_a(ctxb, m0, k2, N_HID);
    a1c = load_a(ctxb, m0 + 16, k2, N_HID);
#pragma unroll
    for (int t = 0; t < 4; t++) bbc[t] = load_b(woutT, n0 + t * 16, k2, N_HID);
#pragma unroll
    for (int t = 0; t < 4; t++) {
      acc[t]     = wmma_bf16(a0n, bbn[t], acc[t]);
      acc[4 + t] = wmma_bf16(a1n, bbn[t], acc[4 + t]);
    }
  }

  int lane = threadIdx.x & 31;
  int nIdx = lane & 15, half = lane >> 4;
#pragma unroll
  for (int mm = 0; mm < 2; mm++) {
#pragma unroll
    for (int t = 0; t < 4; t++) {
#pragma unroll
      for (int r = 0; r < 8; r++) {
        int row = m0 + mm * 16 + r + 8 * half;
        int col = n0 + t * 16 + nIdx;
        size_t idx = (size_t)row * N_HID + col;
        out[idx] = acc[mm * 4 + t][r] + bout[col] + x[idx];
      }
    }
  }
}

// ---------------------------------------------------------------------------
extern "C" void kernel_launch(void* const* d_in, const int* in_sizes, int n_in,
                              void* d_out, int out_size, void* d_ws, size_t ws_size,
                              hipStream_t stream) {
  const float* x    = (const float*)d_in[0];
  const int*   mask = (const int*)d_in[1];
  const float* Wqkv = (const float*)d_in[2];
  const float* bqkv = (const float*)d_in[3];
  const float* Wout = (const float*)d_in[4];
  const float* bout = (const float*)d_in[5];
  float* out = (float*)d_out;

  // workspace carve (all bf16 buffers, 16B-aligned sizes); total ~48 MiB
  char* ws = (char*)d_ws;
  unsigned short* xb     = (unsigned short*)ws; ws += (size_t)ROWS * N_HID * 2;
  unsigned short* wqkvT  = (unsigned short*)ws; ws += (size_t)QKV_N * N_HID * 2;
  unsigned short* qbuf   = (unsigned short*)ws; ws += (size_t)BATCH * N_HEAD * SEQ * HD * 2;
  unsigned short* kbuf   = (unsigned short*)ws; ws += (size_t)BATCH * N_HEAD * SEQ * HD * 2;
  unsigned short* vTbuf  = (unsigned short*)ws; ws += (size_t)BATCH * N_HEAD * HD * SEQ * 2;
  unsigned short* ctxb   = (unsigned short*)ws; ws += (size_t)ROWS * N_HID * 2;
  unsigned short* woutT  = (unsigned short*)ws; ws += (size_t)N_HID * N_HID * 2;

  // prep
  {
    int n = ROWS * N_HID;
    to_bf16_kernel<<<(n + 255) / 256, 256, 0, stream>>>(x, xb, n);
  }
  {
    int n = N_HID * QKV_N;
    transpose_bf16_kernel<<<(n + 255) / 256, 256, 0, stream>>>(Wqkv, wqkvT, N_HID, QKV_N);
  }
  {
    int n = N_HID * N_HID;
    transpose_bf16_kernel<<<(n + 255) / 256, 256, 0, stream>>>(Wout, woutT, N_HID, N_HID);
  }

  // qkv: (4096/32)*(3072/64) = 6144 wave-tiles, 4 waves/block
  qkv_kernel<<<(ROWS / 32) * (QKV_N / 64) / 4, 128, 0, stream>>>(
      xb, wqkvT, bqkv, qbuf, kbuf, vTbuf);

  // attention: 32 (b,h) * 128 q-tiles = 4096 waves
  attn_kernel<<<BATCH * N_HEAD * (SEQ / 16) / 4, 128, 0, stream>>>(
      qbuf, kbuf, vTbuf, mask, ctxb);

  // out projection: (4096/32)*(1024/64) = 2048 waves
  out_kernel<<<(ROWS / 32) * (N_HID / 64) / 4, 128, 0, stream>>>(
      ctxb, woutT, bout, x, out);
}
